// HGT_56152402427945
// MI455X (gfx1250) — compile-verified, matrix operand order checked by reading, not run on hardware
//
#include <hip/hip_runtime.h>
#include <math.h>

// ---------------------------------------------------------------------------
// HGT forward for MI455X (gfx1250, wave32). Dense math via
// v_wmma_f32_16x16x32_f16 with LDS-staged f16 weights (global b128 loads,
// ds_load_b128 operand fetch, straight-line unrolled mainloop).
// Edge passes are streaming gathers + f32 atomics.
// Precondition: all GEMM row counts are multiples of 16 (true here).
// ---------------------------------------------------------------------------

typedef __attribute__((ext_vector_type(16))) _Float16 v16h;
typedef __attribute__((ext_vector_type(8)))  _Float16 v8h;
typedef __attribute__((ext_vector_type(8)))  float    v8f;

#define NPAP   100000
#define NAUT   50000
#define NTOTN  150000
#define EEDGE  250000
#define HIDD   128
#define KDIM   128
#define NHEAD  4
#define HDIM   32
#define OUTD   10
#define RSQRTD 0.17677669529663687f   // 1/sqrt(32)
#define LDW    136                    // LDS stride (halves) for W tile: 272B/col
#define LDR    48                     // LDS stride (halves) for rel tile: 96B/row

__device__ __forceinline__ float gelu_exact(float x) {
  return 0.5f * x * (1.0f + erff(x * 0.7071067811865475f));
}

// float atomic max via signed/unsigned integer monotonicity trick
__device__ __forceinline__ void atomicMaxF(float* addr, float v) {
  if (v >= 0.0f) atomicMax((int*)addr, __float_as_int(v));
  else           atomicMin((unsigned int*)addr, __float_as_uint(v));
}

__device__ __forceinline__ v16h pack_a(const float4 a0, const float4 a1,
                                       const float4 a2, const float4 a3) {
  v16h av;
  av[0]  = (_Float16)a0.x; av[1]  = (_Float16)a0.y;
  av[2]  = (_Float16)a0.z; av[3]  = (_Float16)a0.w;
  av[4]  = (_Float16)a1.x; av[5]  = (_Float16)a1.y;
  av[6]  = (_Float16)a1.z; av[7]  = (_Float16)a1.w;
  av[8]  = (_Float16)a2.x; av[9]  = (_Float16)a2.y;
  av[10] = (_Float16)a2.z; av[11] = (_Float16)a2.w;
  av[12] = (_Float16)a3.x; av[13] = (_Float16)a3.y;
  av[14] = (_Float16)a3.z; av[15] = (_Float16)a3.w;
  return av;
}

// ---------------------------------------------------------------------------
// GEMM: C[M,N] = A[M,128] @ W[128,N] + bias, optional post-ReLU and gated
// residual (beta = sigmoid(*skipp)).  NT = N/16 tiles (compile-time).
// Block = 256 threads = 8 waves. W staged into LDS as f16, transposed [col][k]
// (stride LDW=136 halves -> 272B/col -> conflict-free aligned ds_load_b128).
// Each wave owns a 16-row x N slab: one A fragment (4x global b128) feeds NT
// WMMAs per k-step; K loop fully unrolled (4 steps).
// ---------------------------------------------------------------------------
template <int NT>
__global__ void __launch_bounds__(256)
wmma_gemm(const float* __restrict__ A, const float* __restrict__ W,
          const float* __restrict__ bias, float* __restrict__ C,
          const float* __restrict__ resid, const float* __restrict__ skipp,
          int M, int N, int postRelu)
{
  __shared__ _Float16 ldsW[128 * LDW];

  // ---- cooperative stage: W[k][n] (f32, coalesced) -> ldsW[n][k] (f16) ----
  if (NT == 8) {
    // N == 128: 16384 elements, exactly 64 iterations, no remainder
    for (int idx = threadIdx.x; idx < KDIM * 128; idx += 256) {
      const int k = idx >> 7, n = idx & 127;
      ldsW[n * LDW + k] = (_Float16)W[idx];
    }
  } else {
    const int KN = KDIM * N;
    for (int idx = threadIdx.x; idx < KN; idx += 256) {
      const int k = idx / N, n = idx % N;
      ldsW[n * LDW + k] = (_Float16)W[idx];
    }
  }
  __syncthreads();

  const int lane  = threadIdx.x & 31;
  const int wave  = threadIdx.x >> 5;
  const int tm    = blockIdx.x * 8 + wave;        // 16-row tile index
  if (tm * 16 >= M) return;
  const int r16   = lane & 15;
  const int khalf = (lane >> 4) << 3;             // 0 (lanes 0-15) / 8 (16-31)
  const int row   = tm * 16 + r16;                // always < M (M % 16 == 0)

  v8f acc[NT];
#pragma unroll
  for (int i = 0; i < NT; ++i) acc[i] = (v8f){};

  const float* Ap = A + (size_t)row * KDIM + khalf;
#pragma unroll
  for (int kb = 0; kb < KDIM; kb += 32) {
    // A fragment: K groups [kb+khalf, +8) and [kb+16+khalf, +8) -> 4x b128
    const float4 a0 = *(const float4*)(Ap + kb);
    const float4 a1 = *(const float4*)(Ap + kb + 4);
    const float4 a2 = *(const float4*)(Ap + kb + 16);
    const float4 a3 = *(const float4*)(Ap + kb + 20);
    const v16h av = pack_a(a0, a1, a2, a3);

#pragma unroll
    for (int nt = 0; nt < NT; ++nt) {
      const int col = nt * 16 + r16;
      const v8h g0 = *(const v8h*)(&ldsW[col * LDW + kb + khalf]);
      const v8h g1 = *(const v8h*)(&ldsW[col * LDW + kb + 16 + khalf]);
      v16h bv;
#pragma unroll
      for (int i = 0; i < 8; ++i) { bv[i] = g0[i]; bv[i + 8] = g1[i]; }
      acc[nt] = __builtin_amdgcn_wmma_f32_16x16x32_f16(false, av, false, bv,
                                                       (short)0, acc[nt], false, false);
    }
  }

  // ---- epilogue ----
  float beta = 0.0f;
  if (resid) beta = 1.0f / (1.0f + expf(-skipp[0]));
#pragma unroll
  for (int nt = 0; nt < NT; ++nt) {
    const int col = nt * 16 + r16;
    if (NT == 1 && col >= N) continue;            // only out_lin (N=10) ragged
    const float bval = bias ? bias[col] : 0.0f;
#pragma unroll
    for (int r = 0; r < 8; ++r) {
      const int m = tm * 16 + khalf + r;          // C row: vgpr r (+8 upper half)
      float v = acc[nt][r] + bval;
      if (postRelu) v = v > 0.0f ? v : 0.0f;
      if (resid)    v = beta * v + (1.0f - beta) * resid[(size_t)m * N + col];
      C[(size_t)m * N + col] = v;
    }
  }
}

// ---------------------------------------------------------------------------
// Per-node head transform: O[n, h*32+f] = sum_d X[n, h*32+d] * REL[h][d][f].
// Block = 128 threads = 4 waves; REL (H*D*D = 4096 f32) staged into LDS as f16
// transposed [h][f][d] (stride LDR=48 halves -> aligned ds_load_b128).
// One WMMA (K=32) per wave per 16-row x 16-col tile.  M % 16 == 0 required.
// ---------------------------------------------------------------------------
__global__ void __launch_bounds__(128)
rel_transform(const float* __restrict__ X, const float* __restrict__ REL,
              float* __restrict__ O, int M)
{
  __shared__ _Float16 ldsR[128 * LDR];            // [h*32+f][d], 12KB

#pragma unroll 4
  for (int idx = threadIdx.x; idx < NHEAD * HDIM * HDIM; idx += 128) {
    const int h = idx >> 10;
    const int d = (idx >> 5) & 31;
    const int f = idx & 31;
    ldsR[(h * HDIM + f) * LDR + d] = (_Float16)REL[idx];
  }
  __syncthreads();

  const int lane = threadIdx.x & 31;
  const int wave = threadIdx.x >> 5;
  const int tile = blockIdx.x * 4 + wave;
  const int totTiles = (M >> 4) * 8;              // 8 col-tiles of 16 = HID 128
  if (tile >= totTiles) return;
  const int tm = tile >> 3;
  const int tn = tile & 7;
  const int h  = tn >> 1;
  const int cl = ((tn & 1) << 4) + (lane & 15);   // column within head, 0..31
  const int khalf = (lane >> 4) << 3;
  const int row = tm * 16 + (lane & 15);          // always < M

  const float* Xp = X + (size_t)row * HIDD + h * HDIM + khalf;
  const float4 a0 = *(const float4*)(Xp);
  const float4 a1 = *(const float4*)(Xp + 4);
  const float4 a2 = *(const float4*)(Xp + 16);
  const float4 a3 = *(const float4*)(Xp + 20);
  const v16h av = pack_a(a0, a1, a2, a3);

  const v8h g0 = *(const v8h*)(&ldsR[(h * HDIM + cl) * LDR + khalf]);
  const v8h g1 = *(const v8h*)(&ldsR[(h * HDIM + cl) * LDR + 16 + khalf]);
  v16h bv;
#pragma unroll
  for (int i = 0; i < 8; ++i) { bv[i] = g0[i]; bv[i + 8] = g1[i]; }

  v8f acc = {};
  acc = __builtin_amdgcn_wmma_f32_16x16x32_f16(false, av, false, bv,
                                               (short)0, acc, false, false);
#pragma unroll
  for (int r = 0; r < 8; ++r) {
    const int m = tm * 16 + khalf + r;
    O[(size_t)m * HIDD + h * HDIM + cl] = acc[r];
  }
}

// ---------------------------------------------------------------------------
// Elementwise exact GELU (hoisted out of the GEMM inner loop)
// ---------------------------------------------------------------------------
__global__ void gelu_kernel(const float* __restrict__ in, float* __restrict__ out,
                            size_t n)
{
  const size_t i = (size_t)blockIdx.x * blockDim.x + threadIdx.x;
  if (i < n) out[i] = gelu_exact(in[i]);
}

// ---------------------------------------------------------------------------
// Edge pass 1: scores a[e,h] = (q[dst]·k_rel[src]) * p_rel[h]/sqrt(D); seg max
// ---------------------------------------------------------------------------
__global__ void edge_score(const float* __restrict__ Q, const float* __restrict__ KR,
                           const int* __restrict__ src, const int* __restrict__ dst,
                           const float* __restrict__ prel,
                           float* __restrict__ sc, float* __restrict__ amax,
                           int E, int dstOff, int krOff)
{
  const int e = blockIdx.x * blockDim.x + threadIdx.x;
  if (e >= E) return;
  const size_t srow  = (size_t)(src[e] + krOff) * HIDD;
  const int    dnode = dst[e] + dstOff;
  const size_t drow  = (size_t)dnode * HIDD;
#pragma unroll
  for (int h = 0; h < NHEAD; ++h) {
    const float4* qp = (const float4*)(Q + drow + h * HDIM);
    const float4* kp = (const float4*)(KR + srow + h * HDIM);
    float dotv = 0.0f;
#pragma unroll
    for (int i = 0; i < 8; ++i) {
      const float4 qv = qp[i], kv = kp[i];
      dotv += qv.x * kv.x + qv.y * kv.y + qv.z * kv.z + qv.w * kv.w;
    }
    const float a = dotv * prel[h] * RSQRTD;
    sc[(size_t)e * NHEAD + h] = a;
    atomicMaxF(amax + (size_t)dnode * NHEAD + h, a);
  }
}

// ---------------------------------------------------------------------------
// Edge pass 2: aexp = exp(a - max); accumulate softmax denominator
// ---------------------------------------------------------------------------
__global__ void edge_exp(const float* __restrict__ sc, const int* __restrict__ dst,
                         const float* __restrict__ amax, float* __restrict__ aexp,
                         float* __restrict__ den, int E, int dstOff)
{
  const int e = blockIdx.x * blockDim.x + threadIdx.x;
  if (e >= E) return;
  const int dnode = dst[e] + dstOff;
#pragma unroll
  for (int h = 0; h < NHEAD; ++h) {
    const float v = expf(sc[(size_t)e * NHEAD + h] - amax[(size_t)dnode * NHEAD + h]);
    aexp[(size_t)e * NHEAD + h] = v;
    atomicAdd(den + (size_t)dnode * NHEAD + h, v);
  }
}

// ---------------------------------------------------------------------------
// Edge pass 3: agg[dst] += attn * v_rel[src]; one thread per (edge, head)
// ---------------------------------------------------------------------------
__global__ void edge_agg(const float* __restrict__ aexp, const float* __restrict__ den,
                         const float* __restrict__ VR, const int* __restrict__ src,
                         const int* __restrict__ dst, float* __restrict__ agg,
                         int E, int dstOff, int krOff)
{
  const int t = blockIdx.x * blockDim.x + threadIdx.x;
  if (t >= E * NHEAD) return;
  const int e = t >> 2;
  const int h = t & 3;
  const int dnode = dst[e] + dstOff;
  const float w = aexp[(size_t)e * NHEAD + h] / den[(size_t)dnode * NHEAD + h];
  const float* vp = VR + (size_t)(src[e] + krOff) * HIDD + h * HDIM;
  float* ap = agg + (size_t)dnode * HIDD + h * HDIM;
#pragma unroll
  for (int i = 0; i < HDIM; ++i) atomicAdd(ap + i, vp[i] * w);
}

__global__ void fill_kernel(float* __restrict__ p, float v, size_t n) {
  const size_t i = (size_t)blockIdx.x * blockDim.x + threadIdx.x;
  if (i < n) p[i] = v;
}

// ---------------------------------------------------------------------------
// Host orchestration
// ---------------------------------------------------------------------------
static inline void launch_gemm(const float* A, const float* W, const float* b, float* C,
                               const float* resid, const float* skipp,
                               int M, int N, int postRelu, hipStream_t s)
{
  const int blocks = (M / 16 + 7) / 8;
  if (N == 128)
    wmma_gemm<8><<<blocks, 256, 0, s>>>(A, W, b, C, resid, skipp, M, N, postRelu);
  else
    wmma_gemm<1><<<blocks, 256, 0, s>>>(A, W, b, C, resid, skipp, M, N, postRelu);
}

static inline void launch_rel(const float* X, const float* REL, float* O, int M, hipStream_t s)
{
  const int tiles  = (M / 16) * 8;
  const int blocks = (tiles + 3) / 4;
  rel_transform<<<blocks, 128, 0, s>>>(X, REL, O, M);
}

static inline void launch_fill(float* p, float v, size_t n, hipStream_t s)
{
  fill_kernel<<<(int)((n + 255) / 256), 256, 0, s>>>(p, v, n);
}

extern "C" void kernel_launch(void* const* d_in, const int* in_sizes, int n_in,
                              void* d_out, int out_size, void* d_ws, size_t ws_size,
                              hipStream_t stream)
{
  (void)in_sizes; (void)n_in; (void)out_size; (void)ws_size;

  const float* x_paper  = (const float*)d_in[0];
  const float* x_author = (const float*)d_in[1];
  // edge order matches reference EDGE_TYPES: cites, writes, rev_writes
  const int* e_src[3] = { (const int*)d_in[2], (const int*)d_in[4], (const int*)d_in[6] };
  const int* e_dst[3] = { (const int*)d_in[3], (const int*)d_in[5], (const int*)d_in[7] };
  const float* P = (const float*)d_in[8];
  float* out = (float*)d_out;

  // ---- workspace carve-up (floats) ----
  float* ws = (float*)d_ws;
  size_t off = 0;
  const size_t NREL = 2 * (size_t)NPAP + NAUT;            // 250000 transformed rows
  float* X    = ws + off; off += (size_t)NTOTN * HIDD;
  float* XN   = ws + off; off += (size_t)NTOTN * HIDD;
  float* Kb   = ws + off; off += (size_t)NTOTN * HIDD;    // also gelu scratch
  float* Qb   = ws + off; off += (size_t)NTOTN * HIDD;
  float* Vb   = ws + off; off += (size_t)NTOTN * HIDD;
  float* KR   = ws + off; off += NREL * HIDD;
  float* VR   = ws + off; off += NREL * HIDD;
  float* SC   = ws + off; off += 3 * (size_t)EEDGE * NHEAD;
  float* AEXP = ws + off; off += 3 * (size_t)EEDGE * NHEAD;
  float* AMAX = ws + off; off += (size_t)NTOTN * NHEAD;
  float* DEN  = ws + off; off += (size_t)NTOTN * NHEAD;
  float* AGG  = ws + off; off += (size_t)NTOTN * HIDD;

  // ---- param offsets (jax tree-flatten: dict keys sorted) ----
  // in_lin: author(w,b), paper(w,b)
  const int IN_AUT_W = 0,     IN_AUT_B = 16384;
  const int IN_PAP_W = 16512, IN_PAP_B = 32896;
  const int L0    = 33024;
  const int LSTR  = 156686;   // a(33024)+k(33024)+q(33024)+rel(24588)+skip(2)+v(33024)
  // within-layer offsets; per (w,b) pair b = w + 16384
  const int A_AUT = 0,      A_PAP = 16512;
  const int K_AUT = 33024,  K_PAP = 49536;
  const int Q_AUT = 66048,  Q_PAP = 82560;
  // rel stored sorted: cites, rev_writes, writes; reindex to edge order {cites,writes,rev_writes}
  const int REL_AREL[3] = { 99072, 115464, 107268 };
  const int REL_MREL[3] = { 103168, 119560, 111364 };
  const int REL_PREL[3] = { 107264, 123656, 115460 };
  const int SKIP_AUT = 123660, SKIP_PAP = 123661;
  const int V_AUT = 123662, V_PAP = 140174;
  const int OUT_W = L0 + 2 * LSTR;          // 346396
  const int OUT_B = OUT_W + HIDD * OUTD;    // 347676

  // edge-type tables (edge order: cites, writes, rev_writes)
  const int dstOff[3]  = { 0, 0, NPAP };           // global seg offset of dst type
  const int krOff[3]   = { 0, NPAP, NPAP + NAUT }; // row offset in KR/VR
  const int srcRow[3]  = { 0, NPAP, 0 };           // source rows in combined K/V
  const int srcM[3]    = { NPAP, NAUT, NPAP };

  // ---- input projection: x = relu(x @ Win + b) ----
  launch_gemm(x_paper,  P + IN_PAP_W, P + IN_PAP_B, X,
              nullptr, nullptr, NPAP, HIDD, 1, stream);
  launch_gemm(x_author, P + IN_AUT_W, P + IN_AUT_B, X + (size_t)NPAP * HIDD,
              nullptr, nullptr, NAUT, HIDD, 1, stream);

  for (int l = 0; l < 2; ++l) {
    const float* cur = (l == 0) ? X : XN;
    float*       nxt = (l == 0) ? XN : X;
    const size_t Lb  = (size_t)L0 + (size_t)l * LSTR;

    // K/Q/V projections (per node type)
    launch_gemm(cur, P + Lb + K_PAP, P + Lb + K_PAP + 16384, Kb,
                nullptr, nullptr, NPAP, HIDD, 0, stream);
    launch_gemm(cur + (size_t)NPAP * HIDD, P + Lb + K_AUT, P + Lb + K_AUT + 16384,
                Kb + (size_t)NPAP * HIDD, nullptr, nullptr, NAUT, HIDD, 0, stream);
    launch_gemm(cur, P + Lb + Q_PAP, P + Lb + Q_PAP + 16384, Qb,
                nullptr, nullptr, NPAP, HIDD, 0, stream);
    launch_gemm(cur + (size_t)NPAP * HIDD, P + Lb + Q_AUT, P + Lb + Q_AUT + 16384,
                Qb + (size_t)NPAP * HIDD, nullptr, nullptr, NAUT, HIDD, 0, stream);
    launch_gemm(cur, P + Lb + V_PAP, P + Lb + V_PAP + 16384, Vb,
                nullptr, nullptr, NPAP, HIDD, 0, stream);
    launch_gemm(cur + (size_t)NPAP * HIDD, P + Lb + V_AUT, P + Lb + V_AUT + 16384,
                Vb + (size_t)NPAP * HIDD, nullptr, nullptr, NAUT, HIDD, 0, stream);

    // per-NODE rel transforms (k_rel, v_rel precomputed once per source node)
    for (int et = 0; et < 3; ++et) {
      launch_rel(Kb + (size_t)srcRow[et] * HIDD, P + Lb + REL_AREL[et],
                 KR + (size_t)krOff[et] * HIDD, srcM[et], stream);
      launch_rel(Vb + (size_t)srcRow[et] * HIDD, P + Lb + REL_MREL[et],
                 VR + (size_t)krOff[et] * HIDD, srcM[et], stream);
    }

    // segment-softmax state shared across all edge types
    launch_fill(AMAX, -INFINITY, (size_t)NTOTN * NHEAD, stream);
    launch_fill(DEN,  0.0f,      (size_t)NTOTN * NHEAD, stream);
    launch_fill(AGG,  0.0f,      (size_t)NTOTN * HIDD,  stream);

    const int eb = (EEDGE + 255) / 256;
    for (int et = 0; et < 3; ++et)
      edge_score<<<eb, 256, 0, stream>>>(Qb, KR, e_src[et], e_dst[et],
                                         P + Lb + REL_PREL[et],
                                         SC + (size_t)et * EEDGE * NHEAD, AMAX,
                                         EEDGE, dstOff[et], krOff[et]);
    for (int et = 0; et < 3; ++et)
      edge_exp<<<eb, 256, 0, stream>>>(SC + (size_t)et * EEDGE * NHEAD, e_dst[et],
                                       AMAX, AEXP + (size_t)et * EEDGE * NHEAD,
                                       DEN, EEDGE, dstOff[et]);
    const int ab = (EEDGE * NHEAD + 255) / 256;
    for (int et = 0; et < 3; ++et)
      edge_agg<<<ab, 256, 0, stream>>>(AEXP + (size_t)et * EEDGE * NHEAD, DEN, VR,
                                       e_src[et], e_dst[et], AGG,
                                       EEDGE, dstOff[et], krOff[et]);

    // node update: x' = beta * (gelu(agg) @ Wa + b) + (1-beta) * x
    // GELU hoisted to one streaming pass (Kb reused as scratch here)
    gelu_kernel<<<(int)(((size_t)NTOTN * HIDD + 255) / 256), 256, 0, stream>>>(
        AGG, Kb, (size_t)NTOTN * HIDD);
    launch_gemm(Kb, P + Lb + A_PAP, P + Lb + A_PAP + 16384, nxt,
                cur, P + Lb + SKIP_PAP, NPAP, HIDD, 0, stream);
    launch_gemm(Kb + (size_t)NPAP * HIDD, P + Lb + A_AUT, P + Lb + A_AUT + 16384,
                nxt + (size_t)NPAP * HIDD, cur + (size_t)NPAP * HIDD,
                P + Lb + SKIP_AUT, NAUT, HIDD, 0, stream);
  }

  // output head: papers only, [NP,128] @ [128,10] + b
  launch_gemm(X, P + OUT_W, P + OUT_B, out, nullptr, nullptr,
              NPAP, OUTD, 0, stream);
}
